// ImuNet_7206955123406
// MI455X (gfx1250) — matrix operand order repaired and checked by reading, compiled, not run on hardware
//
#include <hip/hip_runtime.h>
#include <math.h>

typedef _Float16 f16;
typedef __attribute__((ext_vector_type(16))) _Float16 v16h;
typedef __attribute__((ext_vector_type(8)))  _Float16 v8h;
typedef __attribute__((ext_vector_type(8)))  float    v8f;

#define HDIM   512
#define BT     2048
#define NSTEP  11
#define DIN    6
#define K1PAD  32
#define N4H    (4 * HDIM)

// ---------------------------------------------------------------------------
// Prep kernels
// ---------------------------------------------------------------------------

__global__ void zero_kernel(unsigned int* __restrict__ p, int n) {
    int i = blockIdx.x * 256 + threadIdx.x;
    if (i < n) p[i] = 0u;
}

// W: [K][4H] fp32 (row-major, K rows)  ->  Wt: [4H][Kpad] f16, zero-padded K
__global__ void wt_kernel(const float* __restrict__ W, f16* __restrict__ Wt,
                          int K, int Kpad) {
    int i = blockIdx.x * 256 + threadIdx.x;
    if (i >= N4H * Kpad) return;
    int n = i / Kpad;
    int k = i - n * Kpad;
    float v = (k < K) ? W[(size_t)k * N4H + n] : 0.0f;
    Wt[i] = (f16)v;
}

// imus: [B,t,N,D] fp32 with row = b*t+tt  ->  xp: [N][BT][K1PAD] f16 (zero pad)
__global__ void packx_kernel(const float* __restrict__ imus, f16* __restrict__ xp) {
    int i = blockIdx.x * 256 + threadIdx.x;
    if (i >= NSTEP * BT * K1PAD) return;
    int k   = i % K1PAD;
    int row = (i / K1PAD) % BT;
    int n   = i / (K1PAD * BT);
    float v = (k < DIN) ? imus[((size_t)row * NSTEP + n) * DIN + k] : 0.0f;
    xp[i] = (f16)v;
}

// ---------------------------------------------------------------------------
// Fused GEMM(4 gates, 2 M-tiles per wave) + LSTM cell update
// ---------------------------------------------------------------------------

// A fragment (16x32 f16): lane holds row `row`; K = kg*8+[0..7] and 16+kg*8+[0..7]
__device__ __forceinline__ v16h load_a(const f16* __restrict__ A, int K,
                                       int row, int kb, int kg) {
    const v8h* ap = (const v8h*)(A + (size_t)row * K + kb + kg * 8);
    v8h lo = ap[0];
    v8h hi = ap[2];   // +16 f16 elements
    return __builtin_shufflevector(lo, hi, 0, 1, 2, 3, 4, 5, 6, 7,
                                           8, 9, 10, 11, 12, 13, 14, 15);
}

// acc[gate][mtile]: 8 accumulator tiles; each B fragment feeds 2 WMMAs.
__device__ __forceinline__ void gemm_panel(const f16* __restrict__ A, int K,
                                           const f16* __restrict__ Wt,
                                           int mrow0, int ncol, int kg,
                                           v8f (&acc)[4][2]) {
    const size_t gstride = (size_t)HDIM * K;   // gate offset between rows of Wt
    for (int kb = 0; kb < K; kb += 32) {
        v16h a0 = load_a(A, K, mrow0,      kb, kg);
        v16h a1 = load_a(A, K, mrow0 + 16, kb, kg);
        // B fragment: lane holds column ncol (+gate offset); K = kg*8+[0..7] and 16+kg*8+[0..7]
        const int kw = kb + kg * 8;
        const f16* wb = Wt + (size_t)ncol * K + kw;
#pragma unroll
        for (int g = 0; g < 4; ++g) {
            const v8h* bp = (const v8h*)(wb + (size_t)g * gstride);
            v16h b = __builtin_shufflevector(bp[0], bp[2], 0, 1, 2, 3, 4, 5, 6, 7,
                                             8, 9, 10, 11, 12, 13, 14, 15);
            acc[g][0] = __builtin_amdgcn_wmma_f32_16x16x32_f16(
                false, a0, false, b, (short)0, acc[g][0], false, false);
            acc[g][1] = __builtin_amdgcn_wmma_f32_16x16x32_f16(
                false, a1, false, b, (short)0, acc[g][1], false, false);
        }
    }
}

__device__ __forceinline__ float sigf(float x) {
    return 1.0f / (1.0f + __expf(-x));
}

// z = A1@W1 + A2@W2 + b ; split i,f,g,o ; c' = sig(f)*c + sig(i)*tanh(g) ;
// h = sig(o)*tanh(c')
__global__ __launch_bounds__(256)
void lstm_gate_kernel(const f16* __restrict__ A1, int K1,
                      const f16* __restrict__ W1t,   // [4H][K1] f16
                      const f16* __restrict__ A2,    // [BT][H] f16 (h_prev)
                      const f16* __restrict__ W2t,   // [4H][H] f16
                      const float* __restrict__ bias,// [4H] fp32
                      float* __restrict__ c,         // [BT][H] fp32, in/out
                      f16* __restrict__ h_out,       // [BT][H] f16
                      float* __restrict__ h_f32)     // optional [BT][H] fp32
{
    const int lane = threadIdx.x & 31;
    const int wave = threadIdx.x >> 5;
    const int rowBase = blockIdx.x * 32;                 // wave covers 32 rows
    const int colBase = blockIdx.y * 128 + wave * 16;    // 8 waves * 16 cols
    const int mrow0 = rowBase + (lane & 15);
    const int ncol  = colBase + (lane & 15);
    const int kg    = lane >> 4;

    // Init accumulators with bias (broadcast along rows: C element col = ncol)
    v8f acc[4][2];
#pragma unroll
    for (int g = 0; g < 4; ++g) {
        float bv = bias[ncol + g * HDIM];
        v8f s;
#pragma unroll
        for (int e = 0; e < 8; ++e) s[e] = bv;
        acc[g][0] = s;
        acc[g][1] = s;
    }

    gemm_panel(A1, K1,   W1t, mrow0, ncol, kg, acc);
    gemm_panel(A2, HDIM, W2t, mrow0, ncol, kg, acc);

#pragma unroll
    for (int rt = 0; rt < 2; ++rt) {
#pragma unroll
        for (int j = 0; j < 8; ++j) {
            const int row = rowBase + rt * 16 + kg * 8 + j;  // C/D: M = 8*(lane/16)+vgpr
            const size_t idx = (size_t)row * HDIM + ncol;
            float iv = sigf(acc[0][rt][j]);
            float fv = sigf(acc[1][rt][j]);
            float gv = tanhf(acc[2][rt][j]);
            float ov = sigf(acc[3][rt][j]);
            float cn = fv * c[idx] + iv * gv;
            c[idx] = cn;
            float hn = ov * tanhf(cn);
            h_out[idx] = (f16)hn;
            if (h_f32) h_f32[idx] = hn;
        }
    }
}

// ---------------------------------------------------------------------------
// Host launch
// ---------------------------------------------------------------------------

extern "C" void kernel_launch(void* const* d_in, const int* in_sizes, int n_in,
                              void* d_out, int out_size, void* d_ws, size_t ws_size,
                              hipStream_t stream) {
    (void)in_sizes; (void)n_in; (void)out_size; (void)ws_size;
    const float* imus = (const float*)d_in[0];
    const float* k1   = (const float*)d_in[1];
    const float* r1   = (const float*)d_in[2];
    const float* b1   = (const float*)d_in[3];
    const float* k2   = (const float*)d_in[4];
    const float* r2   = (const float*)d_in[5];
    const float* b2   = (const float*)d_in[6];

    char* ws = (char*)d_ws;
    size_t off = 0;
    auto carve = [&](size_t bytes) -> void* {
        void* p = ws + off;
        off = (off + bytes + 255) & ~(size_t)255;
        return p;
    };

    f16* xp   = (f16*)carve((size_t)NSTEP * BT * K1PAD * sizeof(f16));
    f16* k1t  = (f16*)carve((size_t)N4H * K1PAD * sizeof(f16));
    f16* r1t  = (f16*)carve((size_t)N4H * HDIM * sizeof(f16));
    f16* k2t  = (f16*)carve((size_t)N4H * HDIM * sizeof(f16));
    f16* r2t  = (f16*)carve((size_t)N4H * HDIM * sizeof(f16));
    // state block (zeroed in one pass): h1[0], h2[0], c1, c2  -- contiguous
    char* zbase = (char*)(ws + off);
    f16*   h1b0 = (f16*)carve((size_t)BT * HDIM * sizeof(f16));
    f16*   h2b0 = (f16*)carve((size_t)BT * HDIM * sizeof(f16));
    float* c1   = (float*)carve((size_t)BT * HDIM * sizeof(float));
    float* c2   = (float*)carve((size_t)BT * HDIM * sizeof(float));
    size_t zbytes = (size_t)((char*)(ws + off) - zbase);
    f16*   h1b1 = (f16*)carve((size_t)BT * HDIM * sizeof(f16));
    f16*   h2b1 = (f16*)carve((size_t)BT * HDIM * sizeof(f16));

    f16* h1b[2] = {h1b0, h1b1};
    f16* h2b[2] = {h2b0, h2b1};

    // --- prep ---
    {
        int ndw = (int)(zbytes / 4);
        zero_kernel<<<(ndw + 255) / 256, 256, 0, stream>>>((unsigned int*)zbase, ndw);
    }
    {
        int tot = N4H * K1PAD;
        wt_kernel<<<(tot + 255) / 256, 256, 0, stream>>>(k1, k1t, DIN, K1PAD);
        tot = N4H * HDIM;
        wt_kernel<<<(tot + 255) / 256, 256, 0, stream>>>(r1, r1t, HDIM, HDIM);
        wt_kernel<<<(tot + 255) / 256, 256, 0, stream>>>(k2, k2t, HDIM, HDIM);
        wt_kernel<<<(tot + 255) / 256, 256, 0, stream>>>(r2, r2t, HDIM, HDIM);
        tot = NSTEP * BT * K1PAD;
        packx_kernel<<<(tot + 255) / 256, 256, 0, stream>>>(imus, xp);
    }

    // --- recurrent steps ---
    dim3 grid(BT / 32, HDIM / 128);
    int cur = 0;
    for (int n = 0; n < NSTEP; ++n) {
        int nxt = cur ^ 1;
        // layer 1: x_t @ k1 + h1 @ r1
        lstm_gate_kernel<<<grid, 256, 0, stream>>>(
            xp + (size_t)n * BT * K1PAD, K1PAD, k1t,
            h1b[cur], r1t, b1, c1, h1b[nxt], nullptr);
        // layer 2: h1_new @ k2 + h2 @ r2
        float* hfo = (n == NSTEP - 1) ? (float*)d_out : nullptr;
        lstm_gate_kernel<<<grid, 256, 0, stream>>>(
            h1b[nxt], HDIM, k2t,
            h2b[cur], r2t, b2, c2, h2b[nxt], hfo);
        cur = nxt;
    }
}
